// Quantize_15831249453829
// MI455X (gfx1250) — matrix operand order repaired and checked by reading, not compiled, not run on hardware
//
#include <hip/hip_runtime.h>
#include <hip/hip_bf16.h>

typedef __attribute__((ext_vector_type(16))) _Float16 v16h;
typedef __attribute__((ext_vector_type(8)))  float    v8f;

#define DIMK   64
#define NEMB   1024
#define NROWS  65536                 // 64*32*32
#define QELEMS (NROWS * DIMK)        // 4194304
#define IND_OFF   QELEMS
#define DIFF_OFF  (QELEMS + NROWS)
#define ELOSS_OFF (DIFF_OFF + 1)

// ---- zero the two scalar accumulation slots (runs every launch) ----
__global__ void vq_init(float* __restrict__ out) {
    if (threadIdx.x == 0 && blockIdx.x == 0) {
        out[DIFF_OFF]  = 0.0f;
        out[ELOSS_OFF] = 0.0f;
    }
}

// ---- ||E_e||^2 per code ----
__global__ void vq_enorm(const float* __restrict__ embed, float* __restrict__ enorm) {
    int e = blockIdx.x * blockDim.x + threadIdx.x;   // 0..1023
    float s = 0.0f;
#pragma unroll
    for (int k = 0; k < DIMK; ++k) {
        float v = embed[(size_t)k * NEMB + e];
        s += v * v;
    }
    enorm[e] = s;
}

// ---- pack embed (f32, KxN row-major) into f16 WMMA B-operand fragments ----
// Fragment (t, c, lane): 16 halves, element i = B[K = c*32 + (lane>>4)*8 + i + (i>=8 ? 8 : 0)][N = t*16 + (lane&15)]
// Flat index: tid = t*64 + c*32 + lane, dst = wsB + tid*16.
__global__ void vq_packB(const float* __restrict__ embed, _Float16* __restrict__ wsB) {
    int tid  = blockIdx.x * blockDim.x + threadIdx.x;  // 0..4095
    int lane = tid & 31;
    int c    = (tid >> 5) & 1;
    int t    = tid >> 6;
    int col  = t * 16 + (lane & 15);
    int kb   = c * 32 + (lane >> 4) * 8;
    _Float16* dst = wsB + (size_t)tid * 16;
#pragma unroll
    for (int i = 0; i < 16; ++i)
        dst[i] = (_Float16)embed[(size_t)(kb + i + (i >= 8 ? 8 : 0)) * NEMB + col];
}

// ---- main: GEMM (WMMA f16) + argmax + hard-code gather + MSE ----
__global__ __launch_bounds__(128) void vq_main(const float* __restrict__ flat,
                                               const float* __restrict__ embed,
                                               const float* __restrict__ enorm,
                                               const _Float16* __restrict__ wsB,
                                               float* __restrict__ out) {
    __shared__ int sIdx[4 * 16];

    const int lane = threadIdx.x & 31;
    const int wave = threadIdx.x >> 5;
    const int half = lane >> 4;       // lane group 0/1
    const int lr   = lane & 15;
    const int r0   = (blockIdx.x * 4 + wave) * 16;   // 16-row M tile

    // A fragments (row = lr): group0 elems = K{b..b+7, b+16..b+23}, b = half*8
    const float* arow = flat + (size_t)(r0 + lr) * DIMK;
    v16h a0, a1;
    {
        const int b = half * 8;
#pragma unroll
        for (int i = 0; i < 8; ++i) {
            a0[i]     = (_Float16)arow[b + i];
            a0[8 + i] = (_Float16)arow[b + 16 + i];
            a1[i]     = (_Float16)arow[32 + b + i];
            a1[8 + i] = (_Float16)arow[32 + b + 16 + i];
        }
    }

    float best[8];
    int   bidx[8];
#pragma unroll
    for (int j = 0; j < 8; ++j) { best[j] = -3.402823466e+38f; bidx[j] = 0; }

    // Sweep all 64 column tiles; K=64 -> two 16x16x32 WMMAs per tile.
    for (int t = 0; t < 64; ++t) {
        v16h b0 = *(const v16h*)(wsB + ((size_t)t * 64 +      lane) * 16);
        v16h b1 = *(const v16h*)(wsB + ((size_t)t * 64 + 32 + lane) * 16);
        v8f acc = {};
        acc = __builtin_amdgcn_wmma_f32_16x16x32_f16(false, a0, false, b0,
                                                     (short)0, acc, false, false);
        acc = __builtin_amdgcn_wmma_f32_16x16x32_f16(false, a1, false, b1,
                                                     (short)0, acc, false, false);
        const int   col = (t << 4) + lr;
        const float en  = enorm[col];
#pragma unroll
        for (int j = 0; j < 8; ++j) {
            // element j holds (row = half*8 + j, col)
            float s = acc[j] * 2.0f - en;     // argmax(s) == argmin(dist)
            if (s > best[j]) { best[j] = s; bidx[j] = col; }
        }
    }

    // Cross-lane argmax within each 16-lane group (xor masks stay in-group).
#pragma unroll
    for (int off = 8; off >= 1; off >>= 1) {
#pragma unroll
        for (int j = 0; j < 8; ++j) {
            float os = __shfl_xor(best[j], off, 32);
            int   oi = __shfl_xor(bidx[j], off, 32);
            if (os > best[j] || (os == best[j] && oi < bidx[j])) {
                best[j] = os; bidx[j] = oi;
            }
        }
    }

    // Lane 0 / lane 16 publish the 8 row results of their group.
    if (lr == 0) {
#pragma unroll
        for (int j = 0; j < 8; ++j) {
            int row = half * 8 + j;
            sIdx[wave * 16 + row] = bidx[j];
            out[IND_OFF + r0 + row] = (float)bidx[j];   // embed_ind (as float)
        }
    }
    __syncthreads();

    // Hard-code gather + MSE: lane handles row = lane>>1, 32 contiguous dims.
    const int   row = lane >> 1;
    const int   kb  = (lane & 1) * 32;
    const int   e   = sIdx[wave * 16 + row];
    const float* xi = flat + (size_t)(r0 + row) * DIMK + kb;
    float*       qo = out  + (size_t)(r0 + row) * DIMK + kb;
    float sum = 0.0f;
#pragma unroll
    for (int k = 0; k < 32; ++k) {
        float h = embed[(size_t)(kb + k) * NEMB + e];  // L2-resident 256KB table
        float x = xi[k];
        qo[k] = h;                                     // forward value == hard codes
        float d = h - x;
        sum += d * d;
    }
#pragma unroll
    for (int off = 16; off >= 1; off >>= 1) sum += __shfl_xor(sum, off, 32);
    if (lane == 0) {
        const float invN = 1.0f / (float)QELEMS;
        atomicAdd(out + DIFF_OFF,  sum * invN);   // diff
        atomicAdd(out + ELOSS_OFF, sum * invN);   // embed_loss (same value)
    }
}

extern "C" void kernel_launch(void* const* d_in, const int* in_sizes, int n_in,
                              void* d_out, int out_size, void* d_ws, size_t ws_size,
                              hipStream_t stream) {
    const float* inp = (const float*)d_in[0];   // (64,32,32,64) f32
    const float* emb = (const float*)d_in[1];   // (64,1024) f32
    float* out = (float*)d_out;

    float*    enorm = (float*)d_ws;                          // 1024 f32 (4KB)
    _Float16* wsB   = (_Float16*)((char*)d_ws + 4096);       // 65536 f16 (128KB), 32B aligned

    vq_init <<<1,   64,  0, stream>>>(out);
    vq_enorm<<<4,   256, 0, stream>>>(emb, enorm);
    vq_packB<<<16,  256, 0, stream>>>(emb, wsB);
    vq_main <<<1024,128, 0, stream>>>(inp, emb, enorm, wsB, out);
}
